// ContinuousThoughtMachine_40432822124583
// MI455X (gfx1250) — matrix operand order repaired and checked by reading, not compiled, not run on hardware
//
#include <hip/hip_runtime.h>

// Problem dims (match reference)
#define B_    512
#define T_    8
#define D_    2048
#define DIN_  512
#define M_    16
#define H_    4
#define P_    512
#define DOUT_ 512

// GEMM tiling
#define BM 128
#define BN 128
#define BKK 32
#define LDSP 40           // padded bf16 row stride: 64B row + 16B pad (16B-aligned, conflict-free)
#define BUFELEMS (BN * LDSP)

#if defined(__HIP_DEVICE_COMPILE__) && __has_builtin(__builtin_amdgcn_tensor_load_to_lds)
#define USE_TDM 1
#else
#define USE_TDM 0
#endif

typedef __attribute__((ext_vector_type(16))) __bf16 bf16x16;
typedef __attribute__((ext_vector_type(8)))  float  f32x8;
typedef __attribute__((ext_vector_type(4)))  unsigned int u32x4;
typedef __attribute__((ext_vector_type(8)))  int i32x8;
typedef __attribute__((ext_vector_type(4)))  int i32x4;

union FragAB { bf16x16 v; uint4 q[2]; };

__device__ __forceinline__ unsigned short f32_to_bf16(float f) {
  unsigned int u = __float_as_uint(f);
  u += 0x7FFFu + ((u >> 16) & 1u);      // round-to-nearest-even
  return (unsigned short)(u >> 16);
}

// C = A(MxK bf16, row-major, lda) @ Bt(NxK bf16, row-major, ldb)^T + bias
// Optional f32 output (Cf,ldc) and/or bf16 output (Cbf,ldcb).
// B tiles are DMA'd into LDS by the Tensor Data Mover (one 16x32 slice per
// wave, double-buffered), with hardware LDS padding giving the 40-half stride.
__launch_bounds__(256)
__global__ void gemm_bf16_wmma(const unsigned short* __restrict__ A, int lda,
                               const unsigned short* __restrict__ Bt, int ldb,
                               const float* __restrict__ bias,
                               float* __restrict__ Cf, int ldc,
                               unsigned short* __restrict__ Cbf, int ldcb,
                               int K, int Nfull)
{
  __shared__ __align__(128) unsigned short ldsB[2 * BUFELEMS];
  const int tid  = threadIdx.x;
  // wave index is wave-uniform: pin it to an SGPR so the TDM descriptor
  // chain below stays pure-SALU (no per-iteration v_readfirstlane).
  const int wave = __builtin_amdgcn_readfirstlane(tid >> 5);
  const int lane = tid & 31;
  const int rowBase = blockIdx.x * BM + wave * 16;
  const int colBase = blockIdx.y * BN;

  // A fragment addressing: lane L -> row (L&15); K groups {0..7,16..23}+8*(L>>4)
  const int kgrp = lane >> 4;
  const unsigned short* aRowPtr =
      A + (size_t)(rowBase + (lane & 15)) * lda + kgrp * 8;

  const f32x8 zero = {0.f,0.f,0.f,0.f,0.f,0.f,0.f,0.f};
  f32x8 acc[8] = {zero,zero,zero,zero,zero,zero,zero,zero};

  const int nk = K / BKK;

#if USE_TDM
  // --- Tensor Data Mover staging: each wave DMAs its 16-row slice. ---
  const unsigned ldsBase =
      (unsigned)(unsigned long long)(const void*)&ldsB[0] +
      (unsigned)(wave * 16 * LDSP * 2);
  const unsigned long long gSliceBase =
      (unsigned long long)(const void*)(Bt + (size_t)(colBase + wave * 16) * ldb);

  auto tdm_issue = [&](int bufIdx, int kk) {
    unsigned long long ga = gSliceBase + (unsigned long long)kk * 2u;
    u32x4 g0;
    g0.x = 1u;                                   // count=1 (valid user D#)
    g0.y = ldsBase + (unsigned)(bufIdx * BUFELEMS * 2);   // lds_addr (bytes)
    g0.z = (unsigned)(ga & 0xFFFFFFFFull);       // global_addr[31:0]
    g0.w = (unsigned)((ga >> 32) & 0x01FFFFFFull) | (2u << 30); // addr[56:32], type=2
    i32x8 g1;
    // data_size=2B, pad_enable, pad_interval=16 DW (64B), pad_amount=4 DW (16B)
    g1[0] = (1 << 16) | (1 << 20) | (3 << 22) | (3 << 25);
    g1[1] = (int)((unsigned)(K & 0xFFFF) << 16);              // tensor_dim0[15:0]
    g1[2] = (int)(((unsigned)K >> 16) | ((unsigned)(Nfull & 0xFFFF) << 16));
    g1[3] = (int)(((unsigned)Nfull >> 16) | (32u << 16));     // tile_dim0 = 32
    g1[4] = 16;                                  // tile_dim1 = 16, tile_dim2 = 0
    g1[5] = ldb;                                 // tensor_dim0_stride[31:0]
    g1[6] = 0;
    g1[7] = 0;
    i32x4 z4 = {0, 0, 0, 0};
#if __clang_major__ >= 23
    i32x8 z8 = {0, 0, 0, 0, 0, 0, 0, 0};
    __builtin_amdgcn_tensor_load_to_lds(g0, g1, z4, z4, z8, 0);
#else
    __builtin_amdgcn_tensor_load_to_lds(g0, g1, z4, z4, 0);
#endif
  };

  tdm_issue(0, 0);                               // prime buffer 0
  __builtin_amdgcn_s_wait_tensorcnt(0);
  __syncthreads();
#endif

  for (int i = 0; i < nk; ++i) {
    const int kk = i * BKK;
#if USE_TDM
    if (i + 1 < nk) tdm_issue((i + 1) & 1, kk + BKK);   // async next slab
    const unsigned short* ldsTile = &ldsB[(i & 1) * BUFELEMS];
#else
    // Fallback: cooperative global->LDS staging (buffer 0 only).
    {
      const int ldRow = tid >> 1;
      const int ldSeg = (tid & 1) * 16;
      const unsigned short* bp = Bt + (size_t)(colBase + ldRow) * ldb + kk + ldSeg;
      uint4 b0 = *(const uint4*)(bp);
      uint4 b1 = *(const uint4*)(bp + 8);
      __syncthreads();
      *(uint4*)(&ldsB[ldRow * LDSP + ldSeg])     = b0;
      *(uint4*)(&ldsB[ldRow * LDSP + ldSeg] + 8) = b1;
      __syncthreads();
    }
    const unsigned short* ldsTile = &ldsB[0];
#endif

    FragAB a;
    a.q[0] = *(const uint4*)(aRowPtr + kk);        // K = kgrp*8 + 0..7
    a.q[1] = *(const uint4*)(aRowPtr + kk + 16);   // K = kgrp*8 + 16..23
    if (kk + BKK < K)
      __builtin_prefetch((const void*)(aRowPtr + kk + BKK), 0, 1);

    // B fragments: lane L -> column t*16+(L&15), K = (L>>4)*8 + {0..7,16..23}.
    // Two live fragment buffers -> two ds_load pairs in flight, so ds latency
    // hides under the WMMAs instead of an s_wait_dscnt 0 per tile.
    const unsigned short* fragBase = ldsTile + (lane & 15) * LDSP + kgrp * 8;
    FragAB b0, b1;
    b0.q[0] = *(const uint4*)(fragBase);
    b0.q[1] = *(const uint4*)(fragBase + 16);
    b1.q[0] = *(const uint4*)(fragBase + 16 * LDSP);
    b1.q[1] = *(const uint4*)(fragBase + 16 * LDSP + 16);
    #pragma unroll
    for (int t = 0; t < 8; t += 2) {
      acc[t] = __builtin_amdgcn_wmma_f32_16x16x32_bf16(
          false, a.v, false, b0.v, (short)0, acc[t], false, false);
      if (t + 2 < 8) {
        const unsigned short* bp = fragBase + (t + 2) * 16 * LDSP;
        b0.q[0] = *(const uint4*)(bp);
        b0.q[1] = *(const uint4*)(bp + 16);
      }
      acc[t + 1] = __builtin_amdgcn_wmma_f32_16x16x32_bf16(
          false, a.v, false, b1.v, (short)0, acc[t + 1], false, false);
      if (t + 3 < 8) {
        const unsigned short* bp = fragBase + (t + 3) * 16 * LDSP;
        b1.q[0] = *(const uint4*)(bp);
        b1.q[1] = *(const uint4*)(bp + 16);
      }
    }

#if USE_TDM
    if (i + 1 < nk) __builtin_amdgcn_s_wait_tensorcnt(0);
    __syncthreads();
#endif
  }

  // C/D layout: lane L, VGPR v -> row rowBase + (L>>4)*8 + v, col (L&15)
  const int crow = rowBase + kgrp * 8;
  #pragma unroll
  for (int t = 0; t < 8; ++t) {
    const int n = colBase + t * 16 + (lane & 15);
    const float bv = bias ? bias[n] : 0.0f;
    #pragma unroll
    for (int v = 0; v < 8; ++v) {
      const float val = acc[t][v] + bv;
      if (Cf)  Cf [(size_t)(crow + v) * ldc  + n] = val;
      if (Cbf) Cbf[(size_t)(crow + v) * ldcb + n] = f32_to_bf16(val);
    }
  }
}

// in: K x N f32 row-major  ->  out: N x K bf16 row-major
__global__ void transpose_to_bf16(const float* __restrict__ in,
                                  unsigned short* __restrict__ out,
                                  int K, int N)
{
  size_t idx = (size_t)blockIdx.x * 256 + threadIdx.x;
  if (idx >= (size_t)K * N) return;
  int k = (int)(idx % K);
  int n = (int)(idx / K);
  out[idx] = f32_to_bf16(in[(size_t)k * N + n]);
}

__global__ void convert_to_bf16(const float* __restrict__ in,
                                unsigned short* __restrict__ out, int n)
{
  int idx = blockIdx.x * 256 + threadIdx.x;
  if (idx < n) out[idx] = f32_to_bf16(in[idx]);
}

__global__ void zero_f32(float* __restrict__ p, int n)
{
  int idx = blockIdx.x * 256 + threadIdx.x;
  if (idx < n) p[idx] = 0.0f;
}

// z columns of A_cat = broadcast bf16(z0)
__global__ void init_z(const float* __restrict__ z0,
                       unsigned short* __restrict__ Acat)
{
  int idx = blockIdx.x * 256 + threadIdx.x;   // B*D
  int d = idx & (D_ - 1);
  int b = idx >> 11;
  Acat[(size_t)b * (2 * D_) + d] = f32_to_bf16(z0[d]);
}

// Per-neuron MLP over rolling history (slot s holds pre from tick s; weight
// index for pre_s at tick t is M-1-t+s). Writes z (f32) and bf16 z into A_cat.
__global__ void nlm_kernel(const float* __restrict__ hist,   // [T][B*D]
                           const float* __restrict__ w1,     // [M][H][D]
                           const float* __restrict__ b1,     // [H][D]
                           const float* __restrict__ w2,     // [H][D]
                           const float* __restrict__ b2,     // [D]
                           float* __restrict__ z,
                           unsigned short* __restrict__ Acat,
                           int t)
{
  int idx = blockIdx.x * 256 + threadIdx.x;   // B*D
  int d = idx & (D_ - 1);
  float h0 = b1[d], h1 = b1[D_ + d], h2 = b1[2 * D_ + d], h3 = b1[3 * D_ + d];
  for (int s = 0; s <= t; ++s) {
    float p = hist[(size_t)s * (B_ * D_) + idx];
    int m = (M_ - 1 - t) + s;
    const float* wp = w1 + ((size_t)m * H_) * D_ + d;
    h0 = fmaf(p, wp[0],       h0);
    h1 = fmaf(p, wp[D_],      h1);
    h2 = fmaf(p, wp[2 * D_],  h2);
    h3 = fmaf(p, wp[3 * D_],  h3);
  }
  float acc = b2[d];
  acc = fmaf(fmaxf(h0, 0.f), w2[d],          acc);
  acc = fmaf(fmaxf(h1, 0.f), w2[D_ + d],     acc);
  acc = fmaf(fmaxf(h2, 0.f), w2[2 * D_ + d], acc);
  acc = fmaf(fmaxf(h3, 0.f), w2[3 * D_ + d], acc);
  z[idx] = acc;
  int b = idx >> 11;
  Acat[(size_t)b * (2 * D_) + d] = f32_to_bf16(acc);
}

// S_t = exp(-|r|)*S_{t-1} + z_i*z_j for both heads; emit bf16 copies for GEMM.
__global__ void sync_kernel(const float* __restrict__ z,
                            const int* __restrict__ pairs_out,
                            const int* __restrict__ pairs_act,
                            const float* __restrict__ decay_out,
                            const float* __restrict__ decay_act,
                            float* __restrict__ S_out, float* __restrict__ S_act,
                            unsigned short* __restrict__ SoutBf,
                            unsigned short* __restrict__ SactBf)
{
  int idx = blockIdx.x * 256 + threadIdx.x;   // B*P
  int p = idx & (P_ - 1);
  int b = idx >> 9;
  const float* zb = z + (size_t)b * D_;
  {
    int i = pairs_out[2 * p], j = pairs_out[2 * p + 1];
    float r = __expf(-fabsf(decay_out[p]));
    float s = S_out[idx] * r + zb[i] * zb[j];
    S_out[idx] = s;  SoutBf[idx] = f32_to_bf16(s);
  }
  {
    int i = pairs_act[2 * p], j = pairs_act[2 * p + 1];
    float r = __expf(-fabsf(decay_act[p]));
    float s = S_act[idx] * r + zb[i] * zb[j];
    S_act[idx] = s;  SactBf[idx] = f32_to_bf16(s);
  }
}

extern "C" void kernel_launch(void* const* d_in, const int* in_sizes, int n_in,
                              void* d_out, int out_size, void* d_ws, size_t ws_size,
                              hipStream_t stream)
{
  (void)in_sizes; (void)n_in; (void)out_size; (void)ws_size;
  const float* x         = (const float*)d_in[0];
  const float* W_in      = (const float*)d_in[1];
  const float* b_in      = (const float*)d_in[2];
  const float* z0        = (const float*)d_in[3];
  const float* W_syn     = (const float*)d_in[4];
  const float* b_syn     = (const float*)d_in[5];
  const float* nlm_w1    = (const float*)d_in[6];
  const float* nlm_b1    = (const float*)d_in[7];
  const float* nlm_w2    = (const float*)d_in[8];
  const float* nlm_b2    = (const float*)d_in[9];
  const float* decay_out = (const float*)d_in[10];
  const float* decay_act = (const float*)d_in[11];
  const float* W_out     = (const float*)d_in[12];
  const float* b_out     = (const float*)d_in[13];
  const float* W_act     = (const float*)d_in[14];
  const float* b_act     = (const float*)d_in[15];
  const int*   pairs_out = (const int*)d_in[16];
  const int*   pairs_act = (const int*)d_in[17];
  float* out = (float*)d_out;

  // Workspace carve (256B aligned slices), ~66 MB total
  char* w = (char*)d_ws;
  auto carve = [&](size_t bytes) {
    char* p = w; w += (bytes + 255) & ~(size_t)255; return p;
  };
  unsigned short* Acat   = (unsigned short*)carve((size_t)B_ * 2 * D_ * 2);     // [B][2D] bf16: z | feats
  unsigned short* xbf    = (unsigned short*)carve((size_t)B_ * DIN_ * 2);
  unsigned short* WtIn   = (unsigned short*)carve((size_t)D_ * DIN_ * 2);       // D x DIN
  unsigned short* WtSyn  = (unsigned short*)carve((size_t)D_ * 2 * D_ * 2);     // D x 2D
  unsigned short* WtOut  = (unsigned short*)carve((size_t)DOUT_ * P_ * 2);
  unsigned short* WtAct  = (unsigned short*)carve((size_t)DOUT_ * P_ * 2);
  float*          hist   = (float*)carve((size_t)T_ * B_ * D_ * 4);             // rolling pre slots
  float*          zbuf   = (float*)carve((size_t)B_ * D_ * 4);
  float*          Sout   = (float*)carve((size_t)B_ * P_ * 4);
  float*          Sact   = (float*)carve((size_t)B_ * P_ * 4);
  unsigned short* SoutBf = (unsigned short*)carve((size_t)B_ * P_ * 2);
  unsigned short* SactBf = (unsigned short*)carve((size_t)B_ * P_ * 2);

  const dim3 blk(256);

  // ---- prep: bf16 conversions / transposes / state init ----
  transpose_to_bf16<<<(DIN_ * D_ + 255) / 256, blk, 0, stream>>>(W_in,  WtIn,  DIN_,   D_);
  transpose_to_bf16<<<((2 * D_) * D_ + 255) / 256, blk, 0, stream>>>(W_syn, WtSyn, 2 * D_, D_);
  transpose_to_bf16<<<(P_ * DOUT_ + 255) / 256, blk, 0, stream>>>(W_out, WtOut, P_, DOUT_);
  transpose_to_bf16<<<(P_ * DOUT_ + 255) / 256, blk, 0, stream>>>(W_act, WtAct, P_, DOUT_);
  convert_to_bf16<<<(B_ * DIN_ + 255) / 256, blk, 0, stream>>>(x, xbf, B_ * DIN_);
  init_z<<<(B_ * D_) / 256, blk, 0, stream>>>(z0, Acat);
  zero_f32<<<(B_ * P_ + 255) / 256, blk, 0, stream>>>(Sout, B_ * P_);
  zero_f32<<<(B_ * P_ + 255) / 256, blk, 0, stream>>>(Sact, B_ * P_);

  // ---- feats = x @ W_in + b_in -> bf16 into A_cat[:, D:2D] (written once) ----
  gemm_bf16_wmma<<<dim3(B_ / BM, D_ / BN), blk, 0, stream>>>(
      xbf, DIN_, WtIn, DIN_, b_in,
      nullptr, 0, Acat + D_, 2 * D_, DIN_, D_);

  // ---- 8 internal ticks ----
  for (int t = 0; t < T_; ++t) {
    // pre = [z, feats] @ W_syn + b_syn  -> f32 hist slot t
    gemm_bf16_wmma<<<dim3(B_ / BM, D_ / BN), blk, 0, stream>>>(
        Acat, 2 * D_, WtSyn, 2 * D_, b_syn,
        hist + (size_t)t * B_ * D_, D_, nullptr, 0, 2 * D_, D_);

    // z = NLM(hist)  (f32 + bf16 into A_cat[:, 0:D])
    nlm_kernel<<<(B_ * D_) / 256, blk, 0, stream>>>(
        hist, nlm_w1, nlm_b1, nlm_w2, nlm_b2, zbuf, Acat, t);

    // sync recurrence for both heads
    sync_kernel<<<(B_ * P_) / 256, blk, 0, stream>>>(
        zbuf, pairs_out, pairs_act, decay_out, decay_act,
        Sout, Sact, SoutBf, SactBf);

    // ys[:,t,:] = S_out @ W_out + b_out ; qs[:,t,:] = S_act @ W_act + b_act
    gemm_bf16_wmma<<<dim3(B_ / BM, DOUT_ / BN), blk, 0, stream>>>(
        SoutBf, P_, WtOut, P_, b_out,
        out + (size_t)t * DOUT_, T_ * DOUT_, nullptr, 0, P_, DOUT_);
    gemm_bf16_wmma<<<dim3(B_ / BM, DOUT_ / BN), blk, 0, stream>>>(
        SactBf, P_, WtAct, P_, b_act,
        out + (size_t)B_ * T_ * DOUT_ + (size_t)t * DOUT_, T_ * DOUT_, nullptr, 0, P_, DOUT_);
  }
}
